// GCNModel_61718680043839
// MI455X (gfx1250) — matrix operand order repaired and checked by reading, not compiled, not run on hardware
//
#include <hip/hip_runtime.h>
#include <math.h>

typedef __attribute__((ext_vector_type(16))) _Float16 v16h;
typedef __attribute__((ext_vector_type(8)))  _Float16 v8h;
typedef __attribute__((ext_vector_type(8)))  float    v8f;

#define SEQB   12
#define HDIM   300
#define GDIM   500
#define NSPK   9
#define NCLASS 7
#define NLAY   64
#define DA_    1582
#define DV_    342
#define DT_    1024

// padded dims: rows -> multiple of 64 (block tile), K/ld -> multiple of 32 halves
#define PDA   1600   // K-pad of 1582
#define PDV   352    // K-pad of 342
#define PDT   1024   // K-pad of 1024
#define PHK   320    // K-pad of 300
#define PHN   320    // row-pad of 300 (64-mult)
#define PG    512    // pad of 500
#define PG2   1024   // pad of 1000
#define PN3K  1248   // K-pad of 1233
#define PN3M  1280   // row-pad of 1233
#define PNM   448    // row-pad of 411

// ---------------- utility kernels ----------------

__global__ void k_zero16(_Float16* __restrict__ p, int n8) {   // n8 = count of v8h
  int i = blockIdx.x * blockDim.x + threadIdx.x;
  if (i < n8) ((v8h*)p)[i] = (v8h)(_Float16)0.f;
}

__global__ void k_zero(float* __restrict__ p, int n) {
  int i = blockIdx.x * blockDim.x + threadIdx.x;
  if (i < n) p[i] = 0.f;
}

// strided f32 -> f16 convert (row-major -> row-major with dest ld)
__global__ void k_cvt16s(const float* __restrict__ s, int lds,
                         _Float16* __restrict__ d, int ldd, int rows, int cols) {
  int idx = blockIdx.x * blockDim.x + threadIdx.x;
  if (idx >= rows * cols) return;
  int i = idx / cols, c = idx - i * cols;
  d[(size_t)i * ldd + c] = (_Float16)s[(size_t)i * lds + c];
}

// batched transpose-convert: src (L, Ksrc, Nsrc) f32 row-major -> dst (L, Nsrc rows, ldk) f16
__global__ void k_cvtT_b(const float* __restrict__ s, _Float16* __restrict__ d,
                         int L, int Ksrc, int Nsrc, int ldk,
                         size_t sStride, size_t dStride) {
  int idx = blockIdx.x * blockDim.x + threadIdx.x;
  if (idx >= L * Ksrc * Nsrc) return;
  int l = idx / (Ksrc * Nsrc), rem = idx - l * (Ksrc * Nsrc);
  int n = rem / Ksrc, k = rem - n * Ksrc;
  d[dStride * l + (size_t)n * ldk + k] = (_Float16)s[sStride * l + (size_t)k * Nsrc + n];
}

__global__ void k_gather16(const float* __restrict__ src, _Float16* __restrict__ dst,
                           const int* __restrict__ seq, const int* __restrict__ bat,
                           int N, int K, int ldd) {
  int idx = blockIdx.x * blockDim.x + threadIdx.x;
  if (idx >= N * K) return;
  int i = idx / K, k = idx - i * K;
  dst[(size_t)i * ldd + k] = (_Float16)src[((size_t)seq[i] * SEQB + bat[i]) * K + k];
}

__global__ void k_spkadd(float* __restrict__ X, const float* __restrict__ speaker,
                         const float* __restrict__ emb,
                         const int* __restrict__ seq, const int* __restrict__ bat, int N) {
  int idx = blockIdx.x * blockDim.x + threadIdx.x;
  if (idx >= N * HDIM) return;
  int i = idx / HDIM, c = idx - i * HDIM;
  const float* sp = speaker + ((size_t)seq[i] * SEQB + bat[i]) * NSPK;
  int best = 0; float bvv = sp[0];
#pragma unroll
  for (int k = 1; k < NSPK; ++k) { float v = sp[k]; if (v > bvv) { bvv = v; best = k; } }
  X[(size_t)(2 * N + i) * HDIM + c] += emb[best * HDIM + c];
}

__global__ void k_rownorm(const float* __restrict__ X, float* __restrict__ FN,
                          _Float16* __restrict__ FN16) {
  int r = blockIdx.x, lane = threadIdx.x;               // one wave per row
  const float* x = X + (size_t)r * HDIM;
  float s = 0.f;
  for (int c = lane; c < HDIM; c += 32) { float v = x[c]; s += v * v; }
  for (int o = 16; o; o >>= 1) s += __shfl_xor(s, o, 32);
  float inv = 1.f / (sqrtf(s) + 1e-8f);
  for (int c = lane; c < HDIM; c += 32) {
    float v = x[c] * inv;
    FN[(size_t)r * HDIM + c] = v;
    FN16[(size_t)r * PHK + c] = (_Float16)v;
  }
}

__device__ __forceinline__ float arcsim(float c) {
  c *= 0.99999f; c = fminf(1.f, fmaxf(-1.f, c));
  return 1.f - acosf(c) * 0.3183098861837907f;          // 1/pi
}

__global__ void k_xdiag(const float* __restrict__ FN, float* __restrict__ A, int N) {
  int i = blockIdx.x, lane = threadIdx.x;
  const float *f0 = FN + (size_t)i * HDIM,
              *f1 = FN + (size_t)(N + i) * HDIM,
              *f2 = FN + (size_t)(2 * N + i) * HDIM;
  float d01 = 0.f, d02 = 0.f, d12 = 0.f;
  for (int c = lane; c < HDIM; c += 32) {
    float a = f0[c], v = f1[c], t = f2[c];
    d01 += a * v; d02 += a * t; d12 += v * t;
  }
  for (int o = 16; o; o >>= 1) {
    d01 += __shfl_xor(d01, o, 32);
    d02 += __shfl_xor(d02, o, 32);
    d12 += __shfl_xor(d12, o, 32);
  }
  if (lane == 0) {
    int N3 = 3 * N;
    float s01 = arcsim(d01), s02 = arcsim(d02), s12 = arcsim(d12);
    A[(size_t)i * N3 + (N + i)]           = s01;
    A[(size_t)(N + i) * N3 + i]           = s01;
    A[(size_t)i * N3 + (2 * N + i)]       = s02;
    A[(size_t)(2 * N + i) * N3 + i]       = s02;
    A[(size_t)(N + i) * N3 + (2 * N + i)] = s12;
    A[(size_t)(2 * N + i) * N3 + (N + i)] = s12;
  }
}

__global__ void k_rowsum(const float* __restrict__ A, float* __restrict__ deg, int N3) {
  int r = blockIdx.x, lane = threadIdx.x;
  float s = 0.f;
  for (int j = lane; j < N3; j += 32) s += A[(size_t)r * N3 + j];
  for (int o = 16; o; o >>= 1) s += __shfl_xor(s, o, 32);
  if (lane == 0) deg[r] = s;
}

__global__ void k_scale16(const float* __restrict__ A, const float* __restrict__ deg,
                          _Float16* __restrict__ A16, int N3) {
  int idx = blockIdx.x * blockDim.x + threadIdx.x;
  if (idx >= N3 * N3) return;
  int i = idx / N3, j = idx - i * N3;
  float di = deg[i], dj = deg[j];
  float a = (di > 0.f) ? (1.f / sqrtf(di)) : 0.f;
  float b = (dj > 0.f) ? (1.f / sqrtf(dj)) : 0.f;
  A16[(size_t)i * PN3K + j] = (_Float16)(A[idx] * a * b);
}

// ---------------- WMMA GEMM: C = epilogue(A @ Bt^T) ----------------
// A:  Mpad x Kpad f16 row-major (rows multiple of 64, Kpad multiple of 32, padding zeroed)
// Bt: Npad x Kpad f16 row-major (transposed B, same padding guarantees)
// 128 threads = 4 waves; block tile 64x64; each wave owns a 32x32 quadrant (4 WMMAs/K-step).
// Staging is register-pipelined: next K-tile global loads issue before the current
// tile's WMMAs so global latency hides under matrix math.
#define BM 64
#define BN 64
#define BK 32
__global__ void k_wmma_gemm(
    const _Float16* __restrict__ A, int lda,
    const _Float16* __restrict__ Bt, int ldb,
    int M, int Nn, int Kpad,
    float alpha, const float* __restrict__ bias,
    const float* __restrict__ E1, int lde1, float s1,
    const float* __restrict__ E2, int lde2, float s2,
    int relu_f, int adj_mode, const int* __restrict__ dia,
    float* __restrict__ Cf, int ldc,
    _Float16* __restrict__ Ch, int ldch, int transC)
{
  __shared__ _Float16 lsA[BM][BK + 8];
  __shared__ _Float16 lsB[BN][BK + 8];
  const int tid = threadIdx.x;
  const int m0 = blockIdx.y * BM, n0 = blockIdx.x * BN;
  const int wid = tid >> 5, lane = tid & 31;
  const int wy = wid >> 1, wx = wid & 1;
  const int sel = lane >> 4, mr = lane & 15;

  // staging map: thread -> (row sr, 16-half chunk sc); 128 thr * 16 halves = 64x32 tile
  const int sr = tid >> 1, sc = (tid & 1) * 16;
  const v8h* gA = (const v8h*)(A  + (size_t)(m0 + sr) * lda + sc);
  const v8h* gB = (const v8h*)(Bt + (size_t)(n0 + sr) * ldb + sc);
  v8h* sA = (v8h*)&lsA[sr][sc];
  v8h* sB = (v8h*)&lsB[sr][sc];

  v8f acc[2][2] = {};

  // pipeline preload of first K tile
  v8h ra0 = gA[0], ra1 = gA[1], rb0 = gB[0], rb1 = gB[1];
  gA += 4; gB += 4;                                    // +32 halves

  for (int k0 = 0; k0 < Kpad; k0 += BK) {
    // commit staged registers to LDS
    sA[0] = ra0; sA[1] = ra1;
    sB[0] = rb0; sB[1] = rb1;
    __syncthreads();

    // issue next tile's global loads; latency hides under WMMAs below
    if (k0 + BK < Kpad) {
      ra0 = gA[0]; ra1 = gA[1]; rb0 = gB[0]; rb1 = gB[1];
      __builtin_prefetch((const void*)(gA + 4), 0, 1);
      __builtin_prefetch((const void*)(gB + 4), 0, 1);
      gA += 4; gB += 4;
    }

    v16h a[2], b[2];
    {
      const int ar = wy * 32 + mr, kb = sel * 8;
#pragma unroll
      for (int j = 0; j < 8; ++j) {
        a[0][j]     = lsA[ar][kb + j];
        a[0][j + 8] = lsA[ar][kb + 16 + j];
        a[1][j]     = lsA[ar + 16][kb + j];
        a[1][j + 8] = lsA[ar + 16][kb + 16 + j];
      }
      const int br = wx * 32 + mr, kbb = sel * 8;
#pragma unroll
      for (int j = 0; j < 8; ++j) {
        b[0][j]     = lsB[br][kbb + j];
        b[0][j + 8] = lsB[br][kbb + 16 + j];
        b[1][j]     = lsB[br + 16][kbb + j];
        b[1][j + 8] = lsB[br + 16][kbb + 16 + j];
      }
    }
#pragma unroll
    for (int ti = 0; ti < 2; ++ti)
#pragma unroll
      for (int tj = 0; tj < 2; ++tj)
        acc[ti][tj] = __builtin_amdgcn_wmma_f32_16x16x32_f16(
            false, a[ti], false, b[tj], (short)0, acc[ti][tj], false, false);
    __syncthreads();
  }

  // epilogue: C VGPR r, lane l -> M = r + 8*(l>=16), N = l%16
#pragma unroll
  for (int ti = 0; ti < 2; ++ti) {
#pragma unroll
    for (int tj = 0; tj < 2; ++tj) {
      int col = n0 + wx * 32 + tj * 16 + mr;
#pragma unroll
      for (int r = 0; r < 8; ++r) {
        int row = m0 + wy * 32 + ti * 16 + sel * 8 + r;
        if (row < M && col < Nn) {
          float c = acc[ti][tj][r];
          if (adj_mode) {
            c = (dia[row] == dia[col]) ? arcsim(c) : 0.f;
          } else {
            c *= alpha;
            if (bias) c += bias[col];
            if (E1)   c += s1 * E1[(size_t)row * lde1 + col];
            if (E2)   c += s2 * E2[(size_t)row * lde2 + col];
            if (relu_f) c = fmaxf(c, 0.f);
          }
          if (Cf) Cf[(size_t)row * ldc + col] = c;
          if (Ch) {
            if (transC) Ch[(size_t)col * ldch + row] = (_Float16)c;
            else        Ch[(size_t)row * ldch + col] = (_Float16)c;
          }
        }
      }
    }
  }
}

// ---------------- final FC + log_softmax ----------------
__global__ void k_fc_lsm(const float* __restrict__ HF, const float* __restrict__ W,
                         const float* __restrict__ b, float* __restrict__ out, int N) {
  int i = blockIdx.x, lane = threadIdx.x;
  float acc[NCLASS];
#pragma unroll
  for (int c = 0; c < NCLASS; ++c) acc[c] = 0.f;
  for (int j = lane; j < 3 * GDIM; j += 32) {
    int m = j / GDIM, k = j - m * GDIM;
    float v = fmaxf(HF[(size_t)(m * N + i) * GDIM + k], 0.f);
    const float* w = W + (size_t)j * NCLASS;
#pragma unroll
    for (int c = 0; c < NCLASS; ++c) acc[c] += v * w[c];
  }
  for (int o = 16; o; o >>= 1)
#pragma unroll
    for (int c = 0; c < NCLASS; ++c) acc[c] += __shfl_xor(acc[c], o, 32);
  if (lane == 0) {
    float lg[NCLASS], mx = -1e30f;
#pragma unroll
    for (int c = 0; c < NCLASS; ++c) { lg[c] = acc[c] + b[c]; mx = fmaxf(mx, lg[c]); }
    float s = 0.f;
#pragma unroll
    for (int c = 0; c < NCLASS; ++c) s += expf(lg[c] - mx);
    float lse = mx + logf(s);
#pragma unroll
    for (int c = 0; c < NCLASS; ++c) out[i * NCLASS + c] = lg[c] - lse;
  }
}

// ---------------- host ----------------
static inline void* wsal(char*& cur, size_t bytes) {
  void* p = (void*)cur;
  cur += (bytes + 255) & ~(size_t)255;
  return p;
}

extern "C" void kernel_launch(void* const* d_in, const int* in_sizes, int n_in,
                              void* d_out, int out_size, void* d_ws, size_t ws_size,
                              hipStream_t stream) {
  const float* speaker = (const float*)d_in[1];
  const float* fea_a   = (const float*)d_in[2];
  const float* fea_v   = (const float*)d_in[3];
  const float* fea_t   = (const float*)d_in[4];
  const float* Wa = (const float*)d_in[5];  const float* ba = (const float*)d_in[6];
  const float* Wv = (const float*)d_in[7];  const float* bv = (const float*)d_in[8];
  const float* Wt = (const float*)d_in[9];  const float* bt = (const float*)d_in[10];
  const float* spk_emb = (const float*)d_in[11];
  const float* W_in = (const float*)d_in[12]; const float* b_in = (const float*)d_in[13];
  const float* W_convs = (const float*)d_in[14];
  const float* W_fc1 = (const float*)d_in[15]; const float* b_fc1 = (const float*)d_in[16];
  const int* seq_idx = (const int*)d_in[18];
  const int* bat_idx = (const int*)d_in[19];
  const int* dia_id  = (const int*)d_in[20];

  const int N  = in_sizes[18];     // 411
  const int N3 = 3 * N;            // 1233
  (void)ws_size; (void)n_in; (void)out_size;

  // ---- f16 region (contiguous; zeroed once so all padding is 0) ----
  char* cur = (char*)d_ws;
  char* f16_base = cur;
  _Float16* Xa16  = (_Float16*)wsal(cur, (size_t)PNM * PDA * 2);
  _Float16* Xv16  = (_Float16*)wsal(cur, (size_t)PNM * PDV * 2);
  _Float16* Xt16  = (_Float16*)wsal(cur, (size_t)PNM * PDT * 2);
  _Float16* WaT   = (_Float16*)wsal(cur, (size_t)PHN * PDA * 2);   // 300x1582 -> rows n
  _Float16* WvT   = (_Float16*)wsal(cur, (size_t)PHN * PDV * 2);
  _Float16* WtT   = (_Float16*)wsal(cur, (size_t)PHN * PDT * 2);
  _Float16* WinT  = (_Float16*)wsal(cur, (size_t)PG * PHK * 2);    // 500 rows x 300K
  _Float16* WcT   = (_Float16*)wsal(cur, (size_t)NLAY * PG * PG2 * 2); // per layer 512x1024
  _Float16* X16   = (_Float16*)wsal(cur, (size_t)PN3M * PHK * 2);
  _Float16* FN16  = (_Float16*)wsal(cur, (size_t)PN3M * PHK * 2);
  _Float16* A16   = (_Float16*)wsal(cur, (size_t)PN3M * PN3K * 2);
  _Float16* H16T  = (_Float16*)wsal(cur, (size_t)PG * PN3K * 2);   // h^T: 500 rows x 1233K
  _Float16* sup16 = (_Float16*)wsal(cur, (size_t)PN3M * PG2 * 2);  // [hi | h0] rows
  size_t f16_halves = (size_t)(cur - f16_base) / 2;
  // ---- f32 region ----
  float* X    = (float*)wsal(cur, (size_t)N3 * HDIM * 4);
  float* FN   = (float*)wsal(cur, (size_t)N3 * HDIM * 4);
  float* Abig = (float*)wsal(cur, (size_t)N3 * N3 * 4);
  float* degv = (float*)wsal(cur, (size_t)N3 * 4);
  float* H0   = (float*)wsal(cur, (size_t)N3 * GDIM * 4);
  float* HI   = (float*)wsal(cur, (size_t)N3 * GDIM * 4);
  float* HF   = (float*)wsal(cur, (size_t)N3 * GDIM * 4);

  auto ew = [](size_t n) { return dim3((unsigned)((n + 255) / 256)); };

  auto gemm = [&](const _Float16* Ap, int lda, const _Float16* Btp, int ldb,
                  int M, int Nn, int Kpad, float alpha, const float* bias,
                  const float* E1, int lde1, float s1,
                  const float* E2, int lde2, float s2,
                  int relu_f, int adj, const int* dia,
                  float* Cf, int ldc, _Float16* Ch, int ldch, int transC) {
    dim3 g((Nn + BN - 1) / BN, (M + BM - 1) / BM);
    k_wmma_gemm<<<g, dim3(128), 0, stream>>>(Ap, lda, Btp, ldb, M, Nn, Kpad,
                                             alpha, bias, E1, lde1, s1, E2, lde2, s2,
                                             relu_f, adj, dia, Cf, ldc, Ch, ldch, transC);
  };

  // 0) zero the whole f16 workspace (padding must be 0 for the guard-free GEMM)
  k_zero16<<<ew(f16_halves / 8), 256, 0, stream>>>((_Float16*)f16_base, (int)(f16_halves / 8));
  k_zero<<<ew((size_t)N3 * N3), 256, 0, stream>>>(Abig, N3 * N3);

  // 1) weights -> transposed f16
  k_cvtT_b<<<ew((size_t)DA_ * HDIM), 256, 0, stream>>>(Wa, WaT, 1, DA_, HDIM, PDA, 0, 0);
  k_cvtT_b<<<ew((size_t)DV_ * HDIM), 256, 0, stream>>>(Wv, WvT, 1, DV_, HDIM, PDV, 0, 0);
  k_cvtT_b<<<ew((size_t)DT_ * HDIM), 256, 0, stream>>>(Wt, WtT, 1, DT_, HDIM, PDT, 0, 0);
  k_cvtT_b<<<ew((size_t)HDIM * GDIM), 256, 0, stream>>>(W_in, WinT, 1, HDIM, GDIM, PHK, 0, 0);
  k_cvtT_b<<<ew((size_t)NLAY * 2 * GDIM * GDIM), 256, 0, stream>>>(
      W_convs, WcT, NLAY, 2 * GDIM, GDIM, PG2,
      (size_t)2 * GDIM * GDIM, (size_t)PG * PG2);

  // 2) ragged gather + convert
  k_gather16<<<ew((size_t)N * DA_), 256, 0, stream>>>(fea_a, Xa16, seq_idx, bat_idx, N, DA_, PDA);
  k_gather16<<<ew((size_t)N * DV_), 256, 0, stream>>>(fea_v, Xv16, seq_idx, bat_idx, N, DV_, PDV);
  k_gather16<<<ew((size_t)N * DT_), 256, 0, stream>>>(fea_t, Xt16, seq_idx, bat_idx, N, DT_, PDT);

  // 3) projections -> X rows [a | v | t]
  gemm(Xa16, PDA, WaT, PDA, N, HDIM, PDA, 1.f, ba, nullptr, 0, 0.f, nullptr, 0, 0.f,
       0, 0, nullptr, X,                         HDIM, nullptr, 0, 0);
  gemm(Xv16, PDV, WvT, PDV, N, HDIM, PDV, 1.f, bv, nullptr, 0, 0.f, nullptr, 0, 0.f,
       0, 0, nullptr, X + (size_t)N * HDIM,     HDIM, nullptr, 0, 0);
  gemm(Xt16, PDT, WtT, PDT, N, HDIM, PDT, 1.f, bt, nullptr, 0, 0.f, nullptr, 0, 0.f,
       0, 0, nullptr, X + (size_t)2 * N * HDIM, HDIM, nullptr, 0, 0);

  // 4) speaker embedding into text block
  k_spkadd<<<ew((size_t)N * HDIM), 256, 0, stream>>>(X, speaker, spk_emb, seq_idx, bat_idx, N);

  // 5) X -> f16 (padded) for h0 GEMM
  k_cvt16s<<<ew((size_t)N3 * HDIM), 256, 0, stream>>>(X, HDIM, X16, PHK, N3, HDIM);

  // 6) row-normalized features
  k_rownorm<<<N3, 32, 0, stream>>>(X, FN, FN16);

  // 7) adjacency: intra-modal masked arccos-sim + cross-modal diagonals
  for (int m = 0; m < 3; ++m) {
    const _Float16* Fm = FN16 + (size_t)m * N * PHK;
    float* Cblk = Abig + (size_t)(m * N) * N3 + m * N;
    gemm(Fm, PHK, Fm, PHK, N, N, PHK, 1.f, nullptr,
         nullptr, 0, 0.f, nullptr, 0, 0.f, 0, /*adj=*/1, dia_id, Cblk, N3, nullptr, 0, 0);
  }
  k_xdiag<<<N, 32, 0, stream>>>(FN, Abig, N);

  // 8) D^-1/2 A D^-1/2 -> A16 (padded ld)
  k_rowsum<<<N3, 32, 0, stream>>>(Abig, degv, N3);
  k_scale16<<<ew((size_t)N3 * N3), 256, 0, stream>>>(Abig, degv, A16, N3);

  // 9) h0 = relu(X@W_in+b); write f32 H0 and f16 h^T (initial h); h0 into support right half
  gemm(X16, PHK, WinT, PHK, N3, GDIM, PHK, 1.f, b_in,
       nullptr, 0, 0.f, nullptr, 0, 0.f, /*relu=*/1, 0, nullptr,
       H0, GDIM, H16T, PN3K, /*transC=*/1);
  k_cvt16s<<<ew((size_t)N3 * GDIM), 256, 0, stream>>>(H0, GDIM, sup16 + GDIM, PG2, N3, GDIM);

  // 10) 64 GCNII layers
  for (int l = 1; l <= NLAY; ++l) {
    float th = logf(0.5f / (float)l + 1.f);            // LAMDA=0.5
    // hi = Ahat @ h : f32 -> HI, f16 -> support left half (row-major)
    gemm(A16, PN3K, H16T, PN3K, N3, GDIM, PN3K, 1.f, nullptr,
         nullptr, 0, 0.f, nullptr, 0, 0.f, 0, 0, nullptr,
         HI, GDIM, sup16, PG2, /*transC=*/0);
    // h = relu(th*support@W_l + (1-th)*(0.9*hi + 0.1*h0)) ; f16 written transposed
    const _Float16* Wl = WcT + (size_t)(l - 1) * PG * PG2;
    gemm(sup16, PG2, Wl, PG2, N3, GDIM, PG2, th, nullptr,
         HI, GDIM, (1.f - th) * 0.9f, H0, GDIM, (1.f - th) * 0.1f,
         /*relu=*/1, 0, nullptr, HF, GDIM, H16T, PN3K, /*transC=*/1);
  }

  // 11) final FC + log_softmax
  k_fc_lsm<<<N, 32, 0, stream>>>(HF, W_fc1, b_fc1, (float*)d_out, N);
}
